// TorchBoostModel_20701742367032
// MI455X (gfx1250) — matrix operand order repaired and checked by reading, not compiled, not run on hardware
//
#include <hip/hip_runtime.h>
#include <math.h>

#define INPUT_DIM 128
#define HIDDEN    128
#define NUM_TREES 32
#define N_NODES   63
#define N_LEAVES  64
#define ROWS      128     // batch rows per block
#define SH        136     // LDS half-stride (pad to dodge bank conflicts)
#define SHRINK    0.3f
#define BN_EPS    1e-5f

typedef __attribute__((ext_vector_type(16))) _Float16 v16h;
typedef __attribute__((ext_vector_type(8)))  float    v8f;

struct SharedMem {
  _Float16 xA [ROWS      * SH];  // x tile f16, row-major [row][d]
  _Float16 wh [HIDDEN    * SH];  // phase1: w1^T [h][d]; phase2: hidden acts [row][h]
  _Float16 w2t[NUM_TREES * SH];  // w2^T [t][h]
  _Float16 tw [N_LEAVES  * SH];  // current tree weights [node][d]
  float sig[8 * 16 * 64];        // per-wave sigmoid values [wave][row16][node]
  float attnL[ROWS * NUM_TREES]; // attention logits -> softmax probs
  float b1s[HIDDEN], scaleS[HIDDEN], shiftS[HIDDEN];
  float rwS[NUM_TREES], itS[NUM_TREES], b2s[NUM_TREES];
  float tb[N_LEAVES];
  float lv2[2][N_LEAVES];        // double-buffered leaf values
  int   missI[ROWS];
};

// A fragment (16x32 f16): lane l: M=l%16; halfs i=2v+h -> K = kbase + 8*(l/16)
//   + (v&4 ? 16:0) + 2*(v&3) + h   (pairs are contiguous -> dword loads)
__device__ __forceinline__ v16h load_frag_a(const _Float16* base, int row0,
                                            int kbase, int lane) {
  const int m = lane & 15, hi = (lane >> 4) & 1;
  const unsigned* p32 = (const unsigned*)(base + (row0 + m) * SH + kbase + hi * 8);
  union { unsigned u[8]; v16h h; } r;
  r.u[0] = p32[0]; r.u[1] = p32[1]; r.u[2] = p32[2]; r.u[3] = p32[3];
  r.u[4] = p32[8]; r.u[5] = p32[9]; r.u[6] = p32[10]; r.u[7] = p32[11];
  return r.h;
}

// B fragment (32x16 f16): lane l: N=l%16; half i -> K = kbase + 8*(l/16) + (i&8 ? 16:0) + (i&7)
__device__ __forceinline__ v16h load_frag_b(const _Float16* base, int n0,
                                            int kbase, int lane) {
  const int n = lane & 15, hi = (lane >> 4) & 1;
  const unsigned* p32 = (const unsigned*)(base + (n0 + n) * SH + kbase + hi * 8);
  union { unsigned u[8]; v16h h; } r;
#pragma unroll
  for (int i = 0; i < 8; ++i) r.u[i] = p32[(i & 3) + ((i >> 2) << 3)];
  return r.h;
}

__device__ __forceinline__ v8f wmma_f16(v16h a, v16h b, v8f c) {
  return __builtin_amdgcn_wmma_f32_16x16x32_f16(false, a, false, b,
                                                (short)0, c, false, false);
}

__global__ __launch_bounds__(256)
void torchboost_fused(const float* __restrict__ x,
                      const float* __restrict__ tree_w,
                      const float* __restrict__ tree_b,
                      const float* __restrict__ leaf_values,
                      const float* __restrict__ temps,
                      const float* __restrict__ w1,
                      const float* __restrict__ b1,
                      const float* __restrict__ bn_gamma,
                      const float* __restrict__ bn_beta,
                      const float* __restrict__ bn_mean,
                      const float* __restrict__ bn_var,
                      const float* __restrict__ w2,
                      const float* __restrict__ b2,
                      const float* __restrict__ rw,
                      float* __restrict__ out) {
  __shared__ SharedMem sm;
  const int tid  = threadIdx.x;
  const int lane = tid & 31;
  const int wave = tid >> 5;          // 0..7
  const int lm   = lane & 15;
  const int lhi  = lane >> 4;
  const int row0 = blockIdx.x * ROWS; // batch row base of this block
  const int mrow = wave * 16;         // wave's M strip within tile

  // ---------------- stage small vectors / fold BN ----------------
  if (tid < HIDDEN) {
    sm.b1s[tid] = b1[tid];
    float sc = bn_gamma[tid] * rsqrtf(bn_var[tid] + BN_EPS);
    sm.scaleS[tid] = sc;
    sm.shiftS[tid] = bn_beta[tid] - bn_mean[tid] * sc;
  }
  if (tid < NUM_TREES) {
    sm.rwS[tid] = rw[tid];
    sm.itS[tid] = 1.0f / temps[tid];
    sm.b2s[tid] = b2[tid];
  }
  if (tid < ROWS) sm.missI[tid] = 0;
  if (tid < INPUT_DIM) sm.tw[63 * SH + tid] = (_Float16)0.0f; // pad node 63

  // ---------------- stage x tile: f32 -> f16, NaN->0, miss flags ----------
  {
    const float4* xg = (const float4*)(x + (size_t)row0 * INPUT_DIM);
#pragma unroll
    for (int it = 0; it < ROWS * INPUT_DIM / 4 / 256; ++it) {
      int i = tid + it * 256;
      float4 v = xg[i];
      int r = (i * 4) >> 7, c = (i * 4) & 127;
      int m = 0;
      if (v.x != v.x) { v.x = 0.f; m = 1; }
      if (v.y != v.y) { v.y = 0.f; m = 1; }
      if (v.z != v.z) { v.z = 0.f; m = 1; }
      if (v.w != v.w) { v.w = 0.f; m = 1; }
      if (m) atomicOr(&sm.missI[r], 1);
      union { _Float16 h[4]; uint2 u; } t4;
      t4.h[0] = (_Float16)v.x; t4.h[1] = (_Float16)v.y;
      t4.h[2] = (_Float16)v.z; t4.h[3] = (_Float16)v.w;
      *(uint2*)&sm.xA[r * SH + c] = t4.u;
    }
  }
  // stage w1^T f16: coalesced float4 reads, scattered b16 writes
  for (int i = tid; i < HIDDEN * INPUT_DIM / 4; i += 256) {
    int d = i >> 5, h0 = (i & 31) << 2;
    float4 v = *(const float4*)(w1 + d * HIDDEN + h0);
    sm.wh[(h0 + 0) * SH + d] = (_Float16)v.x;
    sm.wh[(h0 + 1) * SH + d] = (_Float16)v.y;
    sm.wh[(h0 + 2) * SH + d] = (_Float16)v.z;
    sm.wh[(h0 + 3) * SH + d] = (_Float16)v.w;
  }
  // stage w2^T f16
  for (int i = tid; i < HIDDEN * NUM_TREES / 4; i += 256) {
    int d = i >> 3, t0 = (i & 7) << 2;
    float4 v = *(const float4*)(w2 + d * NUM_TREES + t0);
    sm.w2t[(t0 + 0) * SH + d] = (_Float16)v.x;
    sm.w2t[(t0 + 1) * SH + d] = (_Float16)v.y;
    sm.w2t[(t0 + 2) * SH + d] = (_Float16)v.z;
    sm.w2t[(t0 + 3) * SH + d] = (_Float16)v.w;
  }
  __syncthreads();

  // x-tile A fragments: invariant across phase 1 AND the whole tree loop
  v16h aF[4];
#pragma unroll
  for (int kt = 0; kt < 4; ++kt) aF[kt] = load_frag_a(sm.xA, mrow, kt * 32, lane);

  // ---------------- phase 1: h = BN(relu(x@w1+b1)) ----------------
  float hreg[8][8];
#pragma unroll
  for (int np = 0; np < 4; ++np) {     // N-tile pairs: interleave 2 accumulators
    v8f a0 = {0.f,0.f,0.f,0.f,0.f,0.f,0.f,0.f};
    v8f a1 = {0.f,0.f,0.f,0.f,0.f,0.f,0.f,0.f};
#pragma unroll
    for (int kt = 0; kt < 4; ++kt) {
      v16h b0 = load_frag_b(sm.wh, (np * 2 + 0) * 16, kt * 32, lane);
      v16h b1 = load_frag_b(sm.wh, (np * 2 + 1) * 16, kt * 32, lane);
      a0 = wmma_f16(aF[kt], b0, a0);
      a1 = wmma_f16(aF[kt], b1, a1);
    }
#pragma unroll
    for (int half = 0; half < 2; ++half) {
      int col = (np * 2 + half) * 16 + lm;
      float bia = sm.b1s[col], sc = sm.scaleS[col], sf = sm.shiftS[col];
#pragma unroll
      for (int r = 0; r < 8; ++r) {
        float hv = fmaxf((half ? a1[r] : a0[r]) + bia, 0.f);
        hreg[np * 2 + half][r] = hv * sc + sf;
      }
    }
  }
  __syncthreads();                     // everyone done reading w1^T
#pragma unroll
  for (int nt = 0; nt < 8; ++nt)
#pragma unroll
    for (int r = 0; r < 8; ++r)
      sm.wh[(mrow + r + lhi * 8) * SH + nt * 16 + lm] = (_Float16)hreg[nt][r];
  __syncthreads();

  // ---------------- phase 2: attn = softmax(h@w2 + b2) ----------------
  {
    v8f a0 = {0.f,0.f,0.f,0.f,0.f,0.f,0.f,0.f};
    v8f a1 = {0.f,0.f,0.f,0.f,0.f,0.f,0.f,0.f};
#pragma unroll
    for (int kt = 0; kt < 4; ++kt) {
      v16h ah = load_frag_a(sm.wh, mrow, kt * 32, lane);
      v16h b0 = load_frag_b(sm.w2t, 0,  kt * 32, lane);
      v16h b1 = load_frag_b(sm.w2t, 16, kt * 32, lane);
      a0 = wmma_f16(ah, b0, a0);
      a1 = wmma_f16(ah, b1, a1);
    }
#pragma unroll
    for (int half = 0; half < 2; ++half) {
      int col = half * 16 + lm;
      float bia = sm.b2s[col];
#pragma unroll
      for (int r = 0; r < 8; ++r)
        sm.attnL[(mrow + r + lhi * 8) * NUM_TREES + col] =
            (half ? a1[r] : a0[r]) + bia;
    }
  }
  __syncthreads();
  if (lane < 16) {                      // per-row softmax over 32 trees
    float* p = &sm.attnL[(mrow + lm) * NUM_TREES];
    float mx = -INFINITY;
    for (int t = 0; t < NUM_TREES; ++t) mx = fmaxf(mx, p[t]);
    float s = 0.f;
    for (int t = 0; t < NUM_TREES; ++t) { float e = __expf(p[t] - mx); p[t] = e; s += e; }
    float inv = 1.f / s;
    for (int t = 0; t < NUM_TREES; ++t) p[t] *= inv;
  }

  int missF[8];
#pragma unroll
  for (int r = 0; r < 8; ++r) missF[r] = sm.missI[mrow + r + lhi * 8];

  // ---------------- phase 3 prologue: stage tree 0 ----------------
  {
    const float4* src = (const float4*)tree_w;
#pragma unroll
    for (int i = 0; i < 8; ++i) {
      int idx = tid + i * 256;
      if (idx < N_NODES * 32) {
        float4 v = src[idx];
        int n = idx >> 5, d = (idx & 31) << 2;
        union { _Float16 h[4]; uint2 u; } t4;
        t4.h[0] = (_Float16)v.x; t4.h[1] = (_Float16)v.y;
        t4.h[2] = (_Float16)v.z; t4.h[3] = (_Float16)v.w;
        *(uint2*)&sm.tw[n * SH + d] = t4.u;
      }
    }
    if (tid < N_LEAVES) {
      sm.tb[tid] = (tid < N_NODES) ? tree_b[tid] : 0.f;
      sm.lv2[0][tid] = leaf_values[tid];
    }
  }
  __syncthreads();

  // ---------------- phase 3: trees (register-pipelined staging) ----------
  float accRow = 0.f;
  for (int t = 0; t < NUM_TREES; ++t) {
    // (1) issue global loads of tree t+1 early (latency hidden behind WMMAs)
    float4 stg[8];
    float tbn = 0.f, lvn = 0.f;
    const bool more = (t + 1 < NUM_TREES);
    if (more) {
      const float4* src =
          (const float4*)(tree_w + (size_t)(t + 1) * N_NODES * INPUT_DIM);
#pragma unroll
      for (int i = 0; i < 8; ++i) {
        int idx = tid + i * 256;
        if (idx < N_NODES * 32) stg[i] = src[idx];
      }
      if (tid < N_LEAVES) {
        tbn = (tid < N_NODES) ? tree_b[(t + 1) * N_NODES + tid] : 0.f;
        lvn = leaf_values[(t + 1) * N_LEAVES + tid];
      }
      if (t + 2 < NUM_TREES) {          // prefetch t+2 (global_prefetch_b8)
        const char* np = (const char*)(tree_w + (size_t)(t + 2) * N_NODES * INPUT_DIM);
        __builtin_prefetch(np + tid * 128, 0, 1);
      }
    }

    // (2) 16 WMMAs on current tree, paired accumulators; sigmoid -> sig
    const float invT = sm.itS[t];
#pragma unroll
    for (int np = 0; np < 2; ++np) {
      v8f a0 = {0.f,0.f,0.f,0.f,0.f,0.f,0.f,0.f};
      v8f a1 = {0.f,0.f,0.f,0.f,0.f,0.f,0.f,0.f};
#pragma unroll
      for (int kt = 0; kt < 4; ++kt) {
        v16h b0 = load_frag_b(sm.tw, (np * 2 + 0) * 16, kt * 32, lane);
        v16h b1 = load_frag_b(sm.tw, (np * 2 + 1) * 16, kt * 32, lane);
        a0 = wmma_f16(aF[kt], b0, a0);
        a1 = wmma_f16(aF[kt], b1, a1);
      }
#pragma unroll
      for (int half = 0; half < 2; ++half) {
        int node = (np * 2 + half) * 16 + lm;
        float bia = sm.tb[node];
#pragma unroll
        for (int r = 0; r < 8; ++r) {
          float z = ((half ? a1[r] : a0[r]) + bia) * invT;
          float s = 1.f / (1.f + __expf(-z));
          if (missF[r]) s = 0.5f;
          sm.sig[(wave * 16 + r + lhi * 8) * 64 + node] = s;
        }
      }
    }
    __syncthreads();   // sig visible; all waves done reading tw/tb

    // (3) store staged tree t+1 into LDS (overlaps leaf walk below)
    if (more) {
#pragma unroll
      for (int i = 0; i < 8; ++i) {
        int idx = tid + i * 256;
        if (idx < N_NODES * 32) {
          int n = idx >> 5, d = (idx & 31) << 2;
          union { _Float16 h[4]; uint2 u; } t4;
          t4.h[0] = (_Float16)stg[i].x; t4.h[1] = (_Float16)stg[i].y;
          t4.h[2] = (_Float16)stg[i].z; t4.h[3] = (_Float16)stg[i].w;
          *(uint2*)&sm.tw[n * SH + d] = t4.u;
        }
      }
      if (tid < N_LEAVES) {
        sm.tb[tid] = tbn;
        sm.lv2[(t + 1) & 1][tid] = lvn;
      }
    }

    // (4) leaf walk: lane handles row lm, bit0=lhi; node(lvl)=(2^l-1)+(j mod 2^l)
    {
      const float* sg = &sm.sig[(wave * 16 + lm) * 64];
      const float* lv = sm.lv2[t & 1];
      const int b0 = lhi;
      float s0 = sg[0];
      float f0 = b0 ? (1.f - s0) : s0;
      float s1 = sg[1 + b0];
      float part = 0.f;
#pragma unroll
      for (int b1 = 0; b1 < 2; ++b1) {
        float f1 = (b1 ? (1.f - s1) : s1) * f0;
        int i1 = b0 + (b1 << 1);
        float s2 = sg[3 + i1];
#pragma unroll
        for (int b2 = 0; b2 < 2; ++b2) {
          float f2 = (b2 ? (1.f - s2) : s2) * f1;
          int i2 = i1 + (b2 << 2);
          float s3 = sg[7 + i2];
#pragma unroll
          for (int b3 = 0; b3 < 2; ++b3) {
            float f3 = (b3 ? (1.f - s3) : s3) * f2;
            int i3 = i2 + (b3 << 3);
            float s4 = sg[15 + i3];
#pragma unroll
            for (int b4 = 0; b4 < 2; ++b4) {
              float f4 = (b4 ? (1.f - s4) : s4) * f3;
              int i4 = i3 + (b4 << 4);
              float s5 = sg[31 + i4];
#pragma unroll
              for (int b5 = 0; b5 < 2; ++b5) {
                float f5 = (b5 ? (1.f - s5) : s5) * f4;
                part = fmaf(f5, lv[i4 + (b5 << 5)], part);
              }
            }
          }
        }
      }
      part += __shfl_xor(part, 16, 32);  // combine even/odd leaf halves
      float at = sm.attnL[(mrow + lm) * NUM_TREES + t];
      accRow = fmaf(part * sm.rwS[t], at, accRow);
    }
    __syncthreads();   // tw/tb/lv stores visible before next iteration
  }

  if (lane < 16) out[row0 + mrow + lm] = SHRINK * accRow;
}

extern "C" void kernel_launch(void* const* d_in, const int* in_sizes, int n_in,
                              void* d_out, int out_size, void* d_ws, size_t ws_size,
                              hipStream_t stream) {
  const float* x      = (const float*)d_in[0];
  const float* tree_w = (const float*)d_in[1];
  const float* tree_b = (const float*)d_in[2];
  const float* leaf_v = (const float*)d_in[3];
  const float* temps  = (const float*)d_in[4];
  const float* w1     = (const float*)d_in[5];
  const float* b1     = (const float*)d_in[6];
  const float* bn_g   = (const float*)d_in[7];
  const float* bn_b   = (const float*)d_in[8];
  const float* bn_m   = (const float*)d_in[9];
  const float* bn_v   = (const float*)d_in[10];
  const float* w2     = (const float*)d_in[11];
  const float* b2     = (const float*)d_in[12];
  const float* rw     = (const float*)d_in[13];
  float* out = (float*)d_out;

  int batch  = in_sizes[0] / INPUT_DIM;
  int blocks = (batch + ROWS - 1) / ROWS;
  torchboost_fused<<<blocks, 256, 0, stream>>>(x, tree_w, tree_b, leaf_v, temps,
                                               w1, b1, bn_g, bn_b, bn_m, bn_v,
                                               w2, b2, rw, out);
}
